// SSMTS_8607114461288
// MI455X (gfx1250) — matrix operand order repaired and checked by reading, not compiled, not run on hardware
//
#include <hip/hip_runtime.h>

// ---------------------------------------------------------------------------
// Mamba (2-layer) forward for MI455X (gfx1250, wave32, WMMA).
// GEMMs run as bf16 16x16x32 WMMA with f32 accumulation; operands are
// pre-packed into the ISA per-lane fragment layout so the GEMM inner loop is
// b128 loads + v_wmma. Each wave register-blocks a 32x32 output tile
// (2 A-frags x 2 B-frags -> 4 WMMAs per K=32 step) and prefetches the
// fragment stream ahead with global_prefetch_b8.
// ---------------------------------------------------------------------------

typedef __bf16 bf16;
typedef __attribute__((ext_vector_type(16))) __bf16 v16bf;
typedef __attribute__((ext_vector_type(8)))  float  v8f;

#define MB_B      2
#define MB_L      1024
#define MB_BL     (MB_B * MB_L)        // 2048 rows
#define MB_DM     1024                 // d_model
#define MB_DI     2048                 // d_inner
#define MB_DS     16                   // d_state
#define MB_DTR    64                   // dt_rank
#define MB_EPS    1e-5f

// ---------------------------------------------------------------------------
// Pack row-major fp32 A (M x K, row stride lda) into bf16 WMMA A-fragments.
// Fragment record: 32 lanes x 16 bf16 (32B per lane), tile order (mt, kt).
// Lane l (l<16): M = l;  lane l>=16: same M, K offset +8 within each 16-K half.
// VGPR v<4  -> K = 2v,2v+1 (+khalf);  v>=4 -> K = 16+2(v-4),+1 (+khalf).
// ---------------------------------------------------------------------------
__global__ void pack_a_kernel(const float* __restrict__ A, bf16* __restrict__ out,
                              int M, int K, int lda) {
    int tid = blockIdx.x * blockDim.x + threadIdx.x;
    int KT = K >> 5;
    int total = (M >> 4) * KT * 32;
    if (tid >= total) return;
    int lane = tid & 31;
    int tile = tid >> 5;
    int kt = tile % KT;
    int mt = tile / KT;
    int m  = (mt << 4) + (lane & 15);
    int khalf = (lane & 16) ? 8 : 0;
    const float* src = A + (size_t)m * lda + ((size_t)kt << 5);
    bf16* dst = out + (size_t)tid * 16;
#pragma unroll
    for (int v = 0; v < 8; ++v) {
        int kb = ((v < 4) ? (v << 1) : (16 + ((v - 4) << 1))) + khalf;
        dst[2 * v]     = (bf16)src[kb];
        dst[2 * v + 1] = (bf16)src[kb + 1];
    }
}

// ---------------------------------------------------------------------------
// Pack row-major fp32 B (K x N, row stride N) into bf16 WMMA B-fragments.
// Tile order (nt, kt). Lane l: N-col = l&15; lanes 16-31 add K offset +8;
// VGPR v<4 -> K = 2v,2v+1 (+khalf); v>=4 -> K = 16+2(v-4),+1 (+khalf).
// ---------------------------------------------------------------------------
__global__ void pack_b_kernel(const float* __restrict__ B, bf16* __restrict__ out,
                              int K, int N) {
    int tid = blockIdx.x * blockDim.x + threadIdx.x;
    int KT = K >> 5;
    int total = (N >> 4) * KT * 32;
    if (tid >= total) return;
    int lane = tid & 31;
    int tile = tid >> 5;
    int kt = tile % KT;
    int nt = tile / KT;
    int n  = (nt << 4) + (lane & 15);
    int kbase = (kt << 5) + ((lane & 16) ? 8 : 0);
    bf16* dst = out + (size_t)tid * 16;
#pragma unroll
    for (int j = 0; j < 8; ++j) {
        int kk = kbase + ((j < 4) ? (j << 1) : (16 + ((j - 4) << 1)));
        dst[2 * j]     = (bf16)B[(size_t)(kk)     * N + n];
        dst[2 * j + 1] = (bf16)B[(size_t)(kk + 1) * N + n];
    }
}

// ---------------------------------------------------------------------------
// WMMA accumulate helper + epilogue tile store.
// ---------------------------------------------------------------------------
__device__ __forceinline__ v8f wmma_bf16(v16bf a, v16bf b, v8f c) {
    return __builtin_amdgcn_wmma_f32_16x16x32_bf16(
        /*neg_a=*/false, a, /*neg_b=*/false, b,
        /*c_mod=*/(short)0, c, /*reuse_a=*/false, /*reuse_b=*/false);
}

__device__ __forceinline__ void store_tile(float* __restrict__ C, int N,
                                           int row, int col, v8f acc,
                                           float bv, int accumulate) {
#pragma unroll
    for (int v = 0; v < 8; ++v) {
        size_t idx = (size_t)(row + v) * N + col;
        float o = acc[v] + bv;
        if (accumulate) o += C[idx];
        C[idx] = o;
    }
}

// ---------------------------------------------------------------------------
// WMMA GEMM: C(MxN,f32) = A(MxK,bf16 frag) * B(KxN,bf16 frag) [+bias][+C].
// One wave per 32x32 output tile (2x2 of 16x16 WMMA tiles), 8 waves/block.
// M, N multiples of 32; K multiple of 32.
// ---------------------------------------------------------------------------
__global__ void wmma_gemm_kernel(const bf16* __restrict__ Apack,
                                 const bf16* __restrict__ Bpack,
                                 const float* __restrict__ bias,
                                 float* __restrict__ C,
                                 int M, int N, int K, int accumulate) {
    int gtid = blockIdx.x * blockDim.x + threadIdx.x;
    int wave = gtid >> 5;
    int lane = gtid & 31;
    int NT2 = N >> 5;
    int MT2 = M >> 5;
    int mt2 = wave / NT2;
    int nt2 = wave % NT2;
    if (mt2 >= MT2) return;                    // uniform per wave
    int KT = K >> 5;

    // fragment streams: tile-major (mt, kt) for A, (nt, kt) for B
    const v16bf* A0 = (const v16bf*)Apack + ((size_t)(2 * mt2) * KT) * 32 + lane;
    const v16bf* A1 = A0 + (size_t)KT * 32;
    const v16bf* B0 = (const v16bf*)Bpack + ((size_t)(2 * nt2) * KT) * 32 + lane;
    const v16bf* B1 = B0 + (size_t)KT * 32;

    v8f c00 = {}, c01 = {}, c10 = {}, c11 = {};
#pragma unroll 2
    for (int kt = 0; kt < KT; ++kt) {
        size_t o = (size_t)kt * 32;
        v16bf a0 = A0[o];
        v16bf a1 = A1[o];
        v16bf b0 = B0[o];
        v16bf b1 = B1[o];
        if (kt + 4 < KT) {                      // warm the cache 4 K-steps ahead
            size_t p = o + 4 * 32;
            __builtin_prefetch((const void*)(A0 + p), 0, 1);
            __builtin_prefetch((const void*)(A1 + p), 0, 1);
            __builtin_prefetch((const void*)(B0 + p), 0, 1);
            __builtin_prefetch((const void*)(B1 + p), 0, 1);
        }
        c00 = wmma_bf16(a0, b0, c00);
        c01 = wmma_bf16(a0, b1, c01);
        c10 = wmma_bf16(a1, b0, c10);
        c11 = wmma_bf16(a1, b1, c11);
    }

    // C/D layout: lane (0-15|16-31) -> col = lane&15; VGPR v -> row v (+8 hi half)
    int col = (nt2 << 5) + (lane & 15);
    int row = (mt2 << 5) + ((lane & 16) ? 8 : 0);
    float bv0 = bias ? bias[col]      : 0.0f;
    float bv1 = bias ? bias[col + 16] : 0.0f;
    store_tile(C, N, row,      col,      c00, bv0, accumulate);
    store_tile(C, N, row,      col + 16, c01, bv1, accumulate);
    store_tile(C, N, row + 16, col,      c10, bv0, accumulate);
    store_tile(C, N, row + 16, col + 16, c11, bv1, accumulate);
}

// ---------------------------------------------------------------------------
// RMSNorm: one block (256 thr = 8 waves) per row of D elements.
// ---------------------------------------------------------------------------
__global__ void rmsnorm_kernel(const float* __restrict__ x, const float* __restrict__ w,
                               float* __restrict__ out, int D) {
    __shared__ float red[9];
    int row = blockIdx.x;
    const float* xr = x + (size_t)row * D;
    float s = 0.0f;
    for (int i = threadIdx.x; i < D; i += blockDim.x) {
        float v = xr[i];
        s += v * v;
    }
#pragma unroll
    for (int off = 16; off > 0; off >>= 1) s += __shfl_down(s, off, 32);
    if ((threadIdx.x & 31) == 0) red[threadIdx.x >> 5] = s;
    __syncthreads();
    if (threadIdx.x == 0) {
        float t = 0.0f;
        for (int i = 0; i < 8; ++i) t += red[i];
        red[8] = t;
    }
    __syncthreads();
    float scale = rsqrtf(red[8] / (float)D + MB_EPS);
    for (int i = threadIdx.x; i < D; i += blockDim.x)
        out[(size_t)row * D + i] = xr[i] * scale * w[i];
}

// ---------------------------------------------------------------------------
// Depthwise causal conv (kernel 4, left pad 3): xc[b,t,d] = b[d] +
//   sum_k w[d,0,k] * xi[b, t+k-3, d].  xi = first DI columns of xz rows (ld).
// ---------------------------------------------------------------------------
__global__ void dwconv_kernel(const float* __restrict__ xz,
                              const float* __restrict__ w,
                              const float* __restrict__ cb,
                              float* __restrict__ xc,
                              int total, int L, int Dh, int ld) {
    int idx = blockIdx.x * blockDim.x + threadIdx.x;
    if (idx >= total) return;
    int d = idx % Dh;
    int t = (idx / Dh) % L;
    int b = idx / (Dh * L);
    const float* base = xz + (size_t)(b * L) * ld + d;
    float acc = cb[d];
#pragma unroll
    for (int k = 0; k < 4; ++k) {
        int tt = t + k - 3;
        if (tt >= 0) acc += w[d * 4 + k] * base[(size_t)tt * ld];
    }
    xc[idx] = acc;
}

// ---------------------------------------------------------------------------
// delta = softplus(dt_raw + 2*dt_b)  (reference adds dt_proj bias twice);
// u = silu(xc), computed in place over both buffers.
// ---------------------------------------------------------------------------
__global__ void delta_u_kernel(float* __restrict__ dt, const float* __restrict__ dtb,
                               float* __restrict__ xc, int Dh, int total) {
    int i = blockIdx.x * blockDim.x + threadIdx.x;
    if (i >= total) return;
    int d = i % Dh;
    float t = dt[i] + 2.0f * dtb[d];
    dt[i] = (t > 20.0f) ? t : log1pf(__expf(t));
    float v = xc[i];
    xc[i] = v / (1.0f + __expf(-v));
}

// ---------------------------------------------------------------------------
// Selective scan: one thread per (b, d) channel, 16 states in registers,
// sequential over L. Epilogue fuses +u*D and *silu(z).
// ---------------------------------------------------------------------------
__global__ void scan_kernel(const float* __restrict__ delta, const float* __restrict__ u,
                            const float* __restrict__ A_log, const float* __restrict__ Dp,
                            const float* __restrict__ xdbl, const float* __restrict__ xz,
                            float* __restrict__ y,
                            int L, int Dh, int ldxz, int ldxd) {
    int idx = blockIdx.x * blockDim.x + threadIdx.x;
    if (idx >= MB_B * Dh) return;
    int d = idx % Dh;
    int b = idx / Dh;

    float Arow[MB_DS];
#pragma unroll
    for (int n = 0; n < MB_DS; ++n) Arow[n] = -__expf(A_log[(size_t)d * MB_DS + n]);
    float h[MB_DS];
#pragma unroll
    for (int n = 0; n < MB_DS; ++n) h[n] = 0.0f;
    float Dd = Dp[d];

    for (int t = 0; t < L; ++t) {
        size_t r = (size_t)(b * L + t);
        float dl = delta[r * Dh + d];
        float uu = u[r * Dh + d];
        const float* Bt = xdbl + r * ldxd + MB_DTR;        // cols 64..79
        const float* Ct = Bt + MB_DS;                      // cols 80..95
        float yv = 0.0f;
#pragma unroll
        for (int n = 0; n < MB_DS; ++n) {
            float hn = __expf(dl * Arow[n]) * h[n] + dl * Bt[n] * uu;
            h[n] = hn;
            yv += hn * Ct[n];
        }
        float zv = xz[r * ldxz + Dh + d];
        y[r * Dh + d] = (yv + uu * Dd) * (zv / (1.0f + __expf(-zv)));
    }
}

// ---------------------------------------------------------------------------
// Host-side launch helpers
// ---------------------------------------------------------------------------
static inline int cdiv(int a, int b) { return (a + b - 1) / b; }

static void run_pack_a(const float* A, bf16* out, int M, int K, int lda, hipStream_t s) {
    int total = (M / 16) * (K / 32) * 32;
    pack_a_kernel<<<cdiv(total, 256), 256, 0, s>>>(A, out, M, K, lda);
}
static void run_pack_b(const float* B, bf16* out, int K, int N, hipStream_t s) {
    int total = (N / 16) * (K / 32) * 32;
    pack_b_kernel<<<cdiv(total, 256), 256, 0, s>>>(B, out, K, N);
}
static void run_gemm(const bf16* A, const bf16* B, const float* bias, float* C,
                     int M, int N, int K, int acc, hipStream_t s) {
    int tiles = (M / 32) * (N / 32);
    wmma_gemm_kernel<<<cdiv(tiles, 8), 256, 0, s>>>(A, B, bias, C, M, N, K, acc);
}

extern "C" void kernel_launch(void* const* d_in, const int* in_sizes, int n_in,
                              void* d_out, int out_size, void* d_ws, size_t ws_size,
                              hipStream_t stream) {
    (void)in_sizes; (void)n_in; (void)out_size; (void)ws_size;

    const float* x     = (const float*)d_in[0];   // (2,1024,32)
    const float* emb_w = (const float*)d_in[1];   // (32,1024)
    const float* emb_b = (const float*)d_in[2];   // (1024)
    struct LayerP {
        const float *norm_w, *in_proj, *conv_w, *conv_b, *x_proj,
                    *dt_proj_w, *dt_proj_b, *A_log, *D, *out_proj;
    } lp[2];
    for (int i = 0; i < 2; ++i) {
        int base = 3 + i * 10;
        lp[i].norm_w    = (const float*)d_in[base + 0];
        lp[i].in_proj   = (const float*)d_in[base + 1];
        lp[i].conv_w    = (const float*)d_in[base + 2];
        lp[i].conv_b    = (const float*)d_in[base + 3];
        lp[i].x_proj    = (const float*)d_in[base + 4];
        lp[i].dt_proj_w = (const float*)d_in[base + 5];
        lp[i].dt_proj_b = (const float*)d_in[base + 6];
        lp[i].A_log     = (const float*)d_in[base + 7];
        lp[i].D         = (const float*)d_in[base + 8];
        lp[i].out_proj  = (const float*)d_in[base + 9];
    }
    const float* norm_f = (const float*)d_in[23]; // (1024)
    const float* out_w  = (const float*)d_in[24]; // (1024,32)
    float* out = (float*)d_out;                   // (2,1024,32)

    // ---- workspace layout (cursor-allocated, 256B aligned) ----
    char* ws = (char*)d_ws;
    size_t off = 0;
    auto alloc = [&](size_t bytes) -> char* {
        char* p = ws + off;
        off += (bytes + 255) & ~(size_t)255;
        return p;
    };
    float* hbuf  = (float*)alloc((size_t)MB_BL * MB_DM * 4);      // residual stream
    float* xn    = (float*)alloc((size_t)MB_BL * MB_DM * 4);      // rmsnorm out
    bf16*  packA = (bf16*) alloc((size_t)MB_BL * MB_DI * 2);      // max A frag
    bf16*  packB = (bf16*) alloc((size_t)MB_DM * 2 * MB_DI * 2);  // max B frag
    float* xz    = (float*)alloc((size_t)MB_BL * 2 * MB_DI * 4);  // in_proj out
    float* xc    = (float*)alloc((size_t)MB_BL * MB_DI * 4);      // conv out -> u (silu)
    float* xdbl  = (float*)alloc((size_t)MB_BL * 96 * 4);         // x_proj out
    float* dtbuf = (float*)alloc((size_t)MB_BL * MB_DI * 4);      // dt_raw -> delta
    float* ybuf  = (float*)alloc((size_t)MB_BL * MB_DI * 4);      // scan out

    // ---- embedding: h = x @ emb_w + emb_b ----
    run_pack_a(x, packA, MB_BL, 32, 32, stream);
    run_pack_b(emb_w, packB, 32, MB_DM, stream);
    run_gemm(packA, packB, emb_b, hbuf, MB_BL, MB_DM, 32, 0, stream);

    // ---- mamba layers ----
    for (int i = 0; i < 2; ++i) {
        const LayerP& p = lp[i];

        rmsnorm_kernel<<<MB_BL, 256, 0, stream>>>(hbuf, p.norm_w, xn, MB_DM);

        // in_proj: xz = xn @ W_in   (2048 x 1024 x 4096)
        run_pack_a(xn, packA, MB_BL, MB_DM, MB_DM, stream);
        run_pack_b(p.in_proj, packB, MB_DM, 2 * MB_DI, stream);
        run_gemm(packA, packB, nullptr, xz, MB_BL, 2 * MB_DI, MB_DM, 0, stream);

        // depthwise causal conv on xi = xz[:, :DI]
        {
            int total = MB_BL * MB_DI;
            dwconv_kernel<<<cdiv(total, 256), 256, 0, stream>>>(
                xz, p.conv_w, p.conv_b, xc, total, MB_L, MB_DI, 2 * MB_DI);
        }

        // x_proj: xdbl = xc @ W_x   (2048 x 2048 x 96)
        run_pack_a(xc, packA, MB_BL, MB_DI, MB_DI, stream);
        run_pack_b(p.x_proj, packB, MB_DI, 96, stream);
        run_gemm(packA, packB, nullptr, xdbl, MB_BL, 96, MB_DI, 0, stream);

        // dt_proj: dt_raw = xdbl[:, :64] @ W_dt  (2048 x 64 x 2048)
        run_pack_a(xdbl, packA, MB_BL, MB_DTR, 96, stream);
        run_pack_b(p.dt_proj_w, packB, MB_DTR, MB_DI, stream);
        run_gemm(packA, packB, nullptr, dtbuf, MB_BL, MB_DI, MB_DTR, 0, stream);

        // delta = softplus(dt_raw + 2b) in place; u = silu(xc) in place
        {
            int total = MB_BL * MB_DI;
            delta_u_kernel<<<cdiv(total, 256), 256, 0, stream>>>(
                dtbuf, p.dt_proj_b, xc, MB_DI, total);
        }

        // selective scan (sequential over L; parallel over b,d)
        scan_kernel<<<cdiv(MB_B * MB_DI, 256), 256, 0, stream>>>(
            dtbuf, xc, p.A_log, p.D, xdbl, xz, ybuf,
            MB_L, MB_DI, 2 * MB_DI, 96);

        // out_proj + residual: h += y @ W_out  (2048 x 2048 x 1024)
        run_pack_a(ybuf, packA, MB_BL, MB_DI, MB_DI, stream);
        run_pack_b(p.out_proj, packB, MB_DI, MB_DM, stream);
        run_gemm(packA, packB, nullptr, hbuf, MB_BL, MB_DM, MB_DI, 1, stream);
    }

    // ---- final norm + head: out = rmsnorm(h) @ out_w  (2048 x 1024 x 32) ----
    rmsnorm_kernel<<<MB_BL, 256, 0, stream>>>(hbuf, norm_f, xn, MB_DM);
    run_pack_a(xn, packA, MB_BL, MB_DM, MB_DM, stream);
    run_pack_b(out_w, packB, MB_DM, 32, stream);
    run_gemm(packA, packB, nullptr, out, MB_BL, 32, MB_DM, 0, stream);
}
